// Block_11046655885712
// MI455X (gfx1250) — compile-verified, matrix operand order checked by reading, not run on hardware
//
#include <hip/hip_runtime.h>
#include <hip/hip_bf16.h>
#include <math.h>

// ---------------------------------------------------------------------------
// Types for CDNA5 WMMA (wave32, 16x16x32 bf16 -> f32)
// ---------------------------------------------------------------------------
typedef unsigned short u16;
typedef __attribute__((ext_vector_type(16))) __bf16 v16bf;
typedef __attribute__((ext_vector_type(8)))  float  v8f;

union FragBF { uint4 u[2]; v16bf v; };

__device__ __forceinline__ u16 f2bf(float f) {
  unsigned u = __float_as_uint(f);
  u += 0x7FFFu + ((u >> 16) & 1u);      // round-to-nearest-even
  return (u16)(u >> 16);
}

__device__ __forceinline__ v8f wmma_bf16(v16bf a, v16bf b, v8f c) {
  // (neg_a, A, neg_b, B, c_mod, C, reuse_a, reuse_b)
  return __builtin_amdgcn_wmma_f32_16x16x32_bf16(false, a, false, b, (short)0, c,
                                                 false, false);
}

// CDNA5 async global->LDS copy (ASYNCcnt-tracked; bypasses VGPRs).
// vdst = per-lane LDS byte offset, vaddr = per-lane 64-bit global address.
__device__ __forceinline__ void async_ld_b128(unsigned lds_off, const void* gaddr) {
  asm volatile("global_load_async_to_lds_b128 %0, %1, off"
               :: "v"(lds_off), "v"(gaddr)
               : "memory");
}
#define WAIT_ASYNC(n) asm volatile("s_wait_asynccnt " #n ::: "memory")

// 16-bit A/B fragment (16 rows x 32 K) per ISA 7.12.2:
// lane l: row = base + (l&15); K-chunks at (l>>4)*8 and 16+(l>>4)*8
__device__ __forceinline__ v16bf frag_g(const u16* __restrict__ p, int ld,
                                        int row, int col) {
  const int lane = threadIdx.x & 31;
  const u16* q = p + (size_t)(row + (lane & 15)) * ld + col + (lane >> 4) * 8;
  FragBF f;
  f.u[0] = *(const uint4*)(q);
  f.u[1] = *(const uint4*)(q + 16);
  return f.v;
}

__device__ __forceinline__ v16bf frag_lds32(const u16* t, int row) {  // stride 32
  const int lane = threadIdx.x & 31;
  const u16* q = t + (row + (lane & 15)) * 32 + (lane >> 4) * 8;
  FragBF f;
  f.u[0] = *(const uint4*)(q);
  f.u[1] = *(const uint4*)(q + 16);
  return f.v;
}

__device__ __forceinline__ v16bf frag_lds64(const u16* t, int row, int kc) {  // stride 64
  const int lane = threadIdx.x & 31;
  const u16* q = t + (row + (lane & 15)) * 64 + kc + (lane >> 4) * 8;
  FragBF f;
  f.u[0] = *(const uint4*)(q);
  f.u[1] = *(const uint4*)(q + 16);
  return f.v;
}

// Row reductions across the 16 lanes holding one C-matrix row (masks 1,2,4,8
// permute only inside each 16-lane half of the wave32).
__device__ __forceinline__ float red16_max(float v) {
  v = fmaxf(v, __shfl_xor(v, 1, 32));
  v = fmaxf(v, __shfl_xor(v, 2, 32));
  v = fmaxf(v, __shfl_xor(v, 4, 32));
  v = fmaxf(v, __shfl_xor(v, 8, 32));
  return v;
}
__device__ __forceinline__ float red16_sum(float v) {
  v += __shfl_xor(v, 1, 32);
  v += __shfl_xor(v, 2, 32);
  v += __shfl_xor(v, 4, 32);
  v += __shfl_xor(v, 8, 32);
  return v;
}

// ---------------------------------------------------------------------------
// fp32 -> bf16 conversion
// ---------------------------------------------------------------------------
__global__ __launch_bounds__(256) void cvt_f32_bf16(const float* __restrict__ in,
                                                    u16* __restrict__ out, int n) {
  int i = blockIdx.x * 256 + threadIdx.x;
  if (i < n) out[i] = f2bf(in[i]);
}

// ---------------------------------------------------------------------------
// LayerNorm (C = 1024 fixed), gather modes:
//   mode 0: temporal rows from x skipping CLS     (rows = B*(N-1))
//   mode 1: spatial rows from {init_cls | xt_new} (rows = B*T*257)
//   mode 2: direct rows from src0 (xcat)          (rows = B*N)
// ---------------------------------------------------------------------------
__global__ __launch_bounds__(256) void ln_k(const float* __restrict__ src0,
                                            const float* __restrict__ src1,
                                            const float* __restrict__ g,
                                            const float* __restrict__ bia,
                                            u16* __restrict__ out,
                                            int mode, int nm1, int T) {
  const int row = blockIdx.x;
  const float* src;
  if (mode == 0) {
    int bb = row / nm1, rr = row - bb * nm1;
    src = src0 + ((size_t)bb * (nm1 + 1) + 1 + rr) * 1024;
  } else if (mode == 1) {
    int sq = row / 257, i = row - sq * 257;
    int bb = sq / T, t = sq - bb * T;
    src = (i == 0) ? src0 + (size_t)bb * (nm1 + 1) * 1024
                   : src1 + ((size_t)bb * nm1 + (size_t)(i - 1) * T + t) * 1024;
  } else {
    src = src0 + (size_t)row * 1024;
  }
  float xv[4];
  float s1 = 0.f, s2 = 0.f;
#pragma unroll
  for (int k = 0; k < 4; ++k) {
    xv[k] = src[threadIdx.x + k * 256];
    s1 += xv[k];
    s2 += xv[k] * xv[k];
  }
#pragma unroll
  for (int m = 1; m < 32; m <<= 1) {
    s1 += __shfl_xor(s1, m, 32);
    s2 += __shfl_xor(s2, m, 32);
  }
  __shared__ float red[16];
  __shared__ float mv[2];
  const int w = threadIdx.x >> 5;
  if ((threadIdx.x & 31) == 0) { red[w * 2] = s1; red[w * 2 + 1] = s2; }
  __syncthreads();
  if (threadIdx.x == 0) {
    float a = 0.f, c = 0.f;
    for (int i = 0; i < 8; ++i) { a += red[i * 2]; c += red[i * 2 + 1]; }
    float mean = a * (1.0f / 1024.0f);
    float var  = c * (1.0f / 1024.0f) - mean * mean;
    mv[0] = mean;
    mv[1] = rsqrtf(var + 1e-5f);
  }
  __syncthreads();
  const float mean = mv[0], rstd = mv[1];
#pragma unroll
  for (int k = 0; k < 4; ++k) {
    int c = threadIdx.x + k * 256;
    out[(size_t)row * 1024 + c] = f2bf((xv[k] - mean) * rstd * g[c] + bia[c]);
  }
}

// ---------------------------------------------------------------------------
// Generic WMMA GEMM:  out[M,N] = A[M,K](bf16) @ Bw[N,K]^T(bf16) (+bias) (+epi)
// Block tile 128x64, K-slab 64, 8 waves (4x2) each computing 32x32.
// Double-buffered LDS fed by async global->LDS copies (ASYNCcnt).
// epi: 0 bf16  1 bf16 gelu  2 f32  3 f32+resid  4 f32+resid(CLS-skip map)
// Requires: N % 64 == 0, K % 64 == 0.
// ---------------------------------------------------------------------------
__global__ __launch_bounds__(256) void gemm_bf16_k(
    const u16* __restrict__ A, const u16* __restrict__ Bw,
    const float* __restrict__ bias, const float* __restrict__ resid,
    void* __restrict__ outp, int M, int N, int K, int epi, int nm1) {
  __shared__ u16 As[2][128 * 64];   // 2 x 16 KB
  __shared__ u16 Bs[2][64 * 64];    // 2 x  8 KB
  const int tid = threadIdx.x;
  const int lane = tid & 31, warp = tid >> 5;
  const int wm = warp & 3, wn = warp >> 2;
  const int bm = blockIdx.y * 128, bn = blockIdx.x * 64;

  v8f acc00 = {}, acc01 = {}, acc10 = {}, acc11 = {};

  // staging maps (per thread): A: 4 x b128, B: 2 x b128 per 64-wide K-slab
  const int arow = bm + (tid >> 1);
  const int aclmp = (arow < M) ? arow : (M - 1);
  const int acol = (tid & 1) * 32;                  // elems
  const int brow = bn + (tid >> 2);
  const int bcol = (tid & 3) * 16;                  // elems
  const u16* agp = A + (size_t)aclmp * K + acol;
  const u16* bgp = Bw + (size_t)brow * K + bcol;
  const unsigned lA = (unsigned)(size_t)(&As[0][(tid >> 1) * 64 + acol]);
  const unsigned lB = (unsigned)(size_t)(&Bs[0][(tid >> 2) * 64 + bcol]);
  const unsigned bufStrideA = (unsigned)(128 * 64 * sizeof(u16));
  const unsigned bufStrideB = (unsigned)(64 * 64 * sizeof(u16));

  auto issue_stage = [&](int buf, int kb) {
    const u16* ga = agp + kb;
    const u16* gb = bgp + kb;
    unsigned la = lA + (unsigned)buf * bufStrideA;
    unsigned lb = lB + (unsigned)buf * bufStrideB;
#pragma unroll
    for (int k = 0; k < 4; ++k)                    // A tile: 128x64
      async_ld_b128(la + k * 16, ga + k * 8);
#pragma unroll
    for (int k = 0; k < 2; ++k)                    // B tile: 64x64
      async_ld_b128(lb + k * 16, gb + k * 8);
  };

  issue_stage(0, 0);
  const int nk = K >> 6;
  for (int i = 0; i < nk; ++i) {
    const int buf = i & 1;
    if (i + 1 < nk) {
      issue_stage(buf ^ 1, (i + 1) * 64);
      WAIT_ASYNC(0x6);                 // retire stage i (6 newer ops pending)
    } else {
      WAIT_ASYNC(0x0);
    }
    __syncthreads();
#pragma unroll
    for (int kc = 0; kc < 64; kc += 32) {
      v16bf a0 = frag_lds64(&As[buf][0], wm * 32,      kc);
      v16bf a1 = frag_lds64(&As[buf][0], wm * 32 + 16, kc);
      v16bf b0 = frag_lds64(&Bs[buf][0], wn * 32,      kc);
      v16bf b1 = frag_lds64(&Bs[buf][0], wn * 32 + 16, kc);
      acc00 = wmma_bf16(a0, b0, acc00);
      acc01 = wmma_bf16(a0, b1, acc01);
      acc10 = wmma_bf16(a1, b0, acc10);
      acc11 = wmma_bf16(a1, b1, acc11);
    }
    __syncthreads();
  }

  const int r0 = lane & 15, hh = (lane >> 4) * 8;
#pragma unroll
  for (int mi = 0; mi < 2; ++mi) {
#pragma unroll
    for (int ni = 0; ni < 2; ++ni) {
      v8f acc = (mi == 0) ? (ni == 0 ? acc00 : acc01) : (ni == 0 ? acc10 : acc11);
#pragma unroll
      for (int r = 0; r < 8; ++r) {
        int row = bm + wm * 32 + mi * 16 + r + hh;
        int col = bn + wn * 32 + ni * 16 + r0;
        if (row < M) {
          float v = acc[r] + (bias ? bias[col] : 0.0f);
          size_t oidx = (size_t)row * N + col;
          switch (epi) {
            case 0: ((u16*)outp)[oidx] = f2bf(v); break;
            case 1:
              ((u16*)outp)[oidx] =
                  f2bf(0.5f * v * (1.0f + erff(v * 0.70710678118654752f)));
              break;
            case 2: ((float*)outp)[oidx] = v; break;
            case 3: ((float*)outp)[oidx] = v + resid[oidx]; break;
            case 4: {
              int b = row / nm1, rr = row - b * nm1;
              ((float*)outp)[oidx] =
                  v + resid[((size_t)b * (nm1 + 1) + 1 + rr) * N + col];
            } break;
          }
        }
      }
    }
  }
}

// ---------------------------------------------------------------------------
// Temporal attention: seq len 16 == one WMMA tile. One wave per (seq, head).
// qkv layout [M_t, 3072]: q at h*64, k at 1024+h*64, v at 2048+h*64.
// ---------------------------------------------------------------------------
__global__ __launch_bounds__(128) void attn_temporal_k(const u16* __restrict__ qkv,
                                                       u16* __restrict__ outb) {
  __shared__ u16 p_lds[4][16 * 32];
  __shared__ u16 v_lds[4][64 * 32];
  const int warp = threadIdx.x >> 5, lane = threadIdx.x & 31;
  const int unit = blockIdx.x * 4 + warp;
  const int seq = unit >> 4, h = unit & 15;
  const int row0 = seq * 16;
  const int cq = h * 64, ck = 1024 + h * 64, cv = 2048 + h * 64;
  const int r0 = lane & 15, hf = lane >> 4;

  v16bf q0 = frag_g(qkv, 3072, row0, cq);
  v16bf q1 = frag_g(qkv, 3072, row0, cq + 32);
  v16bf k0 = frag_g(qkv, 3072, row0, ck);
  v16bf k1 = frag_g(qkv, 3072, row0, ck + 32);
  v8f s = {};
  s = wmma_bf16(q0, k0, s);
  s = wmma_bf16(q1, k1, s);

  u16* pl = p_lds[warp];
#pragma unroll
  for (int r = 0; r < 8; ++r) {
    float v = s[r] * 0.125f;                 // SCALE = 64^-0.5
    float m = red16_max(v);
    float e = __expf(v - m);
    float su = red16_sum(e);
    pl[(r + 8 * hf) * 32 + r0] = f2bf(e / su);
    pl[(r + 8 * hf) * 32 + r0 + 16] = 0;     // K padding
  }
  // stage V^T (64 d-rows x 32 j-cols, j>=16 zero)
  u16* vt = v_lds[warp];
  for (int dd = lane; dd < 64; dd += 32) {
#pragma unroll
    for (int j = 0; j < 16; ++j)
      vt[dd * 32 + j] = qkv[(size_t)(row0 + j) * 3072 + cv + dd];
#pragma unroll
    for (int j = 16; j < 32; ++j) vt[dd * 32 + j] = 0;
  }
  __syncthreads();

  v16bf pf = frag_lds32(pl, 0);
#pragma unroll
  for (int db = 0; db < 4; ++db) {
    v16bf vf = frag_lds32(vt, db * 16);
    v8f o = {};
    o = wmma_bf16(pf, vf, o);
#pragma unroll
    for (int r = 0; r < 8; ++r) {
      int mrow = row0 + r + 8 * hf;
      outb[(size_t)mrow * 1024 + h * 64 + db * 16 + r0] = f2bf(o[r]);
    }
  }
}

// ---------------------------------------------------------------------------
// Spatial attention (L = 257), flash-style online softmax over 17 key tiles.
// One wave per (seq, head, row-tile). qkv buffer padded by >=16 rows.
// ---------------------------------------------------------------------------
__global__ __launch_bounds__(128) void attn_spatial_k(const u16* __restrict__ qkv,
                                                      u16* __restrict__ outb) {
  __shared__ u16 p_lds[4][16 * 32];
  __shared__ u16 v_lds[4][64 * 32];
  const int warp = threadIdx.x >> 5, lane = threadIdx.x & 31;
  const int unit = blockIdx.x * 4 + warp;
  const int rt = unit % 17;
  const int sh = unit / 17;
  const int h = sh & 15, sq = sh >> 4;
  const int L = 257;
  const size_t base = (size_t)sq * L;
  const int cq = h * 64, ck = 1024 + h * 64, cv = 2048 + h * 64;
  const int r0 = lane & 15, hf = lane >> 4;

  v16bf q0 = frag_g(qkv, 3072, (int)base + rt * 16, cq);
  v16bf q1 = frag_g(qkv, 3072, (int)base + rt * 16, cq + 32);

  float rmax[8], rsum[8];
  v8f o0 = {}, o1 = {}, o2 = {}, o3 = {};
#pragma unroll
  for (int r = 0; r < 8; ++r) { rmax[r] = -1e30f; rsum[r] = 0.f; }

  u16* pl = p_lds[warp];
  u16* vt = v_lds[warp];

  for (int jt = 0; jt < 17; ++jt) {
    const int jrow = jt * 16;
    v16bf k0 = frag_g(qkv, 3072, (int)base + jrow, ck);
    v16bf k1 = frag_g(qkv, 3072, (int)base + jrow, ck + 32);
    v8f sc = {};
    sc = wmma_bf16(q0, k0, sc);
    sc = wmma_bf16(q1, k1, sc);

    const int jg = jrow + r0;
    float fac[8];
#pragma unroll
    for (int r = 0; r < 8; ++r) {
      float v = sc[r] * 0.125f;
      if (jg >= L) v = -1e30f;
      float m = red16_max(v);
      float nm = fmaxf(rmax[r], m);
      float e = __expf(v - nm);
      float su = red16_sum(e);
      fac[r] = __expf(rmax[r] - nm);
      rsum[r] = rsum[r] * fac[r] + su;
      rmax[r] = nm;
      pl[(r + 8 * hf) * 32 + r0] = f2bf(e);
      pl[(r + 8 * hf) * 32 + r0 + 16] = 0;
    }
#pragma unroll
    for (int r = 0; r < 8; ++r) {
      o0[r] *= fac[r]; o1[r] *= fac[r]; o2[r] *= fac[r]; o3[r] *= fac[r];
    }
    for (int dd = lane; dd < 64; dd += 32) {
#pragma unroll
      for (int j = 0; j < 16; ++j) {
        int jj = jrow + j;
        vt[dd * 32 + j] =
            (jj < L) ? qkv[(base + jj) * 3072 + cv + dd] : (u16)0;
      }
#pragma unroll
      for (int j = 16; j < 32; ++j) vt[dd * 32 + j] = 0;
    }
    __syncthreads();
    v16bf pf = frag_lds32(pl, 0);
    o0 = wmma_bf16(pf, frag_lds32(vt, 0),  o0);
    o1 = wmma_bf16(pf, frag_lds32(vt, 16), o1);
    o2 = wmma_bf16(pf, frag_lds32(vt, 32), o2);
    o3 = wmma_bf16(pf, frag_lds32(vt, 48), o3);
    __syncthreads();
  }

#pragma unroll
  for (int r = 0; r < 8; ++r) {
    int i = rt * 16 + r + 8 * hf;
    if (i < L) {
      float inv = 1.0f / rsum[r];
      size_t ob = (base + i) * 1024 + h * 64 + r0;
      outb[ob + 0]  = f2bf(o0[r] * inv);
      outb[ob + 16] = f2bf(o1[r] * inv);
      outb[ob + 32] = f2bf(o2[r] * inv);
      outb[ob + 48] = f2bf(o3[r] * inv);
    }
  }
}

// ---------------------------------------------------------------------------
// xcat = concat(init_cls, xt_new) + concat(mean_t(res_s[:,0]), res_sp)
// One block per output row (B*N rows), 256 threads x 4 cols.
// ---------------------------------------------------------------------------
__global__ __launch_bounds__(256) void combine_k(const float* __restrict__ x,
                                                 const float* __restrict__ xtnew,
                                                 const float* __restrict__ res_s,
                                                 float* __restrict__ out,
                                                 int nm1, int T) {
  const int row = blockIdx.x;
  const int NF = nm1 + 1;
  const int bb = row / NF, n = row - bb * NF;
#pragma unroll
  for (int k = 0; k < 4; ++k) {
    int c = threadIdx.x + k * 256;
    float v;
    if (n == 0) {
      float s = 0.f;
      for (int t = 0; t < 16; ++t)
        s += res_s[((size_t)(bb * T + t) * 257) * 1024 + c];
      v = x[(size_t)bb * NF * 1024 + c] + s * (1.0f / (float)T);
    } else {
      int rr = n - 1;
      int p = rr / T, t = rr - p * T;
      v = xtnew[((size_t)bb * nm1 + rr) * 1024 + c] +
          res_s[((size_t)(bb * T + t) * 257 + 1 + p) * 1024 + c];
    }
    out[(size_t)row * 1024 + c] = v;
  }
}

// ---------------------------------------------------------------------------
// Host orchestration
// ---------------------------------------------------------------------------
extern "C" void kernel_launch(void* const* d_in, const int* in_sizes, int n_in,
                              void* d_out, int out_size, void* d_ws, size_t ws_size,
                              hipStream_t stream) {
  // constants matching setup_inputs()
  constexpr int C = 1024, C3 = 3072, MH = 4096;
  constexpr int Bb = 8, T = 16, NM1 = 4096, NF = 4097;
  constexpr int MT = 32768;        // temporal tokens  B*(N-1)
  constexpr int L = 257, S = Bb * T;
  constexpr int MS = S * L;        // 32896 spatial tokens
  constexpr int MSP = MS + 32;     // padded for flash boundary frags
  constexpr int MA = Bb * NF;      // 32776 total rows

  const float* x       = (const float*)d_in[0];
  const float* norm1_g = (const float*)d_in[1];
  const float* norm1_b = (const float*)d_in[2];
  const float* qkv_w   = (const float*)d_in[3];
  const float* proj_w  = (const float*)d_in[4];
  const float* proj_b  = (const float*)d_in[5];
  const float* tnorm_g = (const float*)d_in[6];
  const float* tnorm_b = (const float*)d_in[7];
  const float* tqkv_w  = (const float*)d_in[8];
  const float* tproj_w = (const float*)d_in[9];
  const float* tproj_b = (const float*)d_in[10];
  const float* tfc_w   = (const float*)d_in[11];
  const float* tfc_b   = (const float*)d_in[12];
  const float* norm2_g = (const float*)d_in[13];
  const float* norm2_b = (const float*)d_in[14];
  const float* fc1_w   = (const float*)d_in[15];
  const float* fc1_b   = (const float*)d_in[16];
  const float* fc2_w   = (const float*)d_in[17];
  const float* fc2_b   = (const float*)d_in[18];
  float* out = (float*)d_out;

  char* ws = (char*)d_ws;
  size_t off = 0;
  auto carve = [&](size_t bytes) -> void* {
    void* p = (void*)(ws + off);
    off += (bytes + 255) & ~(size_t)255;
    return p;
  };
  u16* w_tqkv  = (u16*)carve((size_t)C3 * C * 2);
  u16* w_tproj = (u16*)carve((size_t)C * C * 2);
  u16* w_tfc   = (u16*)carve((size_t)C * C * 2);
  u16* w_qkv   = (u16*)carve((size_t)C3 * C * 2);
  u16* w_proj  = (u16*)carve((size_t)C * C * 2);
  u16* w_fc1   = (u16*)carve((size_t)MH * C * 2);
  u16* w_fc2   = (u16*)carve((size_t)C * MH * 2);
  u16* ln_buf  = (u16*)carve((size_t)MSP * C * 2);
  u16* qkv_buf = (u16*)carve((size_t)MSP * C3 * 2);
  u16* attn_buf= (u16*)carve((size_t)MS * C * 2);
  u16* projt   = (u16*)carve((size_t)MT * C * 2);
  float* xtnew = (float*)carve((size_t)MT * C * 4);
  u16* h1      = (u16*)carve((size_t)MA * MH * 2);
  float* res_s = (float*)qkv_buf;  // alias: qkv_s dead before proj_s writes

  auto cvt = [&](const float* s, u16* d, int n) {
    cvt_f32_bf16<<<dim3((n + 255) / 256), dim3(256), 0, stream>>>(s, d, n);
  };
  auto gemm = [&](const u16* A, const u16* Bw, const float* bias,
                  const float* resid, void* o, int M, int N, int K, int epi) {
    dim3 g(N / 64, (M + 127) / 128);
    gemm_bf16_k<<<g, dim3(256), 0, stream>>>(A, Bw, bias, resid, o, M, N, K,
                                             epi, NM1);
  };

  // weight conversions
  cvt(tqkv_w, w_tqkv, C3 * C);
  cvt(tproj_w, w_tproj, C * C);
  cvt(tfc_w, w_tfc, C * C);
  cvt(qkv_w, w_qkv, C3 * C);
  cvt(proj_w, w_proj, C * C);
  cvt(fc1_w, w_fc1, MH * C);
  cvt(fc2_w, w_fc2, C * MH);

  // ---- temporal branch ----
  ln_k<<<dim3(MT), dim3(256), 0, stream>>>(x, nullptr, tnorm_g, tnorm_b, ln_buf,
                                           0, NM1, T);
  gemm(ln_buf, w_tqkv, nullptr, nullptr, qkv_buf, MT, C3, C, 0);
  attn_temporal_k<<<dim3((MT / 16) * 16 / 4), dim3(128), 0, stream>>>(qkv_buf,
                                                                      attn_buf);
  gemm(attn_buf, w_tproj, tproj_b, nullptr, projt, MT, C, C, 0);
  gemm(projt, w_tfc, tfc_b, x, xtnew, MT, C, C, 4);  // + x[:,1:,:] residual

  // ---- spatial branch ----
  ln_k<<<dim3(MS), dim3(256), 0, stream>>>(x, xtnew, norm1_g, norm1_b, ln_buf,
                                           1, NM1, T);
  gemm(ln_buf, w_qkv, nullptr, nullptr, qkv_buf, MS, C3, C, 0);
  attn_spatial_k<<<dim3(S * 16 * 17 / 4), dim3(128), 0, stream>>>(qkv_buf,
                                                                  attn_buf);
  gemm(attn_buf, w_proj, proj_b, nullptr, res_s, MS, C, C, 2);

  // ---- combine residuals into xcat (stored in d_out) ----
  combine_k<<<dim3(MA), dim3(256), 0, stream>>>(x, xtnew, res_s, out, NM1, T);

  // ---- MLP ----
  ln_k<<<dim3(MA), dim3(256), 0, stream>>>(out, nullptr, norm2_g, norm2_b,
                                           ln_buf, 2, NM1, T);
  gemm(ln_buf, w_fc1, fc1_b, nullptr, h1, MA, MH, C, 1);       // GELU
  gemm(h1, w_fc2, fc2_b, out, out, MA, C, MH, 3);              // + xcat
}
